// PointNetFeaturePropagation_34187939676754
// MI455X (gfx1250) — matrix operand order repaired and checked by reading, not compiled, not run on hardware
//
#include <hip/hip_runtime.h>

// ---------------------------------------------------------------------------
// PointNet FeaturePropagation + TokenEmbed + PosExtraction for MI455X (gfx1250)
// All 1x1-conv GEMMs run on v_wmma_f32_16x16x32_bf16 (f32 accumulate).
// Shapes (fixed by the reference):
//   B=8, N=4096, S=1024, D1=128, D2=256, C0=384, 2*C0=768, OC=512, G=512, n=8
// ---------------------------------------------------------------------------

typedef float  v8f   __attribute__((ext_vector_type(8)));
typedef __bf16 v16bf __attribute__((ext_vector_type(16)));
typedef unsigned int v4u __attribute__((ext_vector_type(4)));

#define BATCH   8
#define NPTS    4096
#define SPTS    1024
#define D1C     128
#define D2C     256
#define C0      384        // in_c
#define C1      768        // 2*in_c
#define OC      512        // out_c
#define GGRP    512
#define NSUB    8          // n = N / G
#define NCOL0   (BATCH*NPTS)   // 32768 columns for TokenEmbed GEMMs
#define NCOLG   (BATCH*GGRP)   // 4096 columns for PosExtraction GEMMs

__device__ __forceinline__ unsigned short f2bf(float f) {
    unsigned int u = __builtin_bit_cast(unsigned int, f);
    unsigned int r = u + 0x7FFFu + ((u >> 16) & 1u);   // round-to-nearest-even
    return (unsigned short)(r >> 16);
}
__device__ __forceinline__ float bf2f(unsigned short h) {
    unsigned int u = ((unsigned int)h) << 16;
    return __builtin_bit_cast(float, u);
}

// ---------------------------------------------------------------------------
// Weight conversion f32 -> bf16 (weights stay L2/WGP$-resident thereafter)
// ---------------------------------------------------------------------------
__global__ void cvt_w_kernel(const float* __restrict__ src,
                             unsigned short* __restrict__ dst, int nelem) {
    int i = blockIdx.x * blockDim.x + threadIdx.x;
    if (i < nelem) dst[i] = f2bf(src[i]);
}

// ---------------------------------------------------------------------------
// 3-NN inverse-distance interpolation + concat(points1, interp) -> X0 bf16
// X0 layout: [col][384] feature-contiguous ("X^T"), col = b*4096 + p.
// ---------------------------------------------------------------------------
__global__ __launch_bounds__(128)
void knn_interp_kernel(const float* __restrict__ xyz1,
                       const float* __restrict__ xyz2,
                       const float* __restrict__ pts1,
                       const float* __restrict__ pts2,
                       unsigned short* __restrict__ X0) {
    __shared__ float sx[SPTS * 3];         // 12 KB of the 320 KB WGP LDS
    const int b = blockIdx.y;
    const int p = blockIdx.x * 128 + threadIdx.x;

    const float* x2b = xyz2 + (size_t)b * SPTS * 3;
    for (int j = threadIdx.x; j < SPTS * 3; j += 128) sx[j] = x2b[j];
    __syncthreads();

    const float* x1 = xyz1 + ((size_t)b * NPTS + p) * 3;
    const float px = x1[0], py = x1[1], pz = x1[2];

    float d0 = 3.0e38f, d1 = 3.0e38f, d2 = 3.0e38f;
    int   i0 = 0, i1 = 0, i2 = 0;
    for (int s = 0; s < SPTS; ++s) {
        float dx = px - sx[s * 3 + 0];
        float dy = py - sx[s * 3 + 1];
        float dz = pz - sx[s * 3 + 2];
        float d  = dx * dx + dy * dy + dz * dz;
        if (d < d2) {
            if (d < d1) {
                if (d < d0) { d2 = d1; i2 = i1; d1 = d0; i1 = i0; d0 = d; i0 = s; }
                else        { d2 = d1; i2 = i1; d1 = d;  i1 = s; }
            } else          { d2 = d;  i2 = s; }
        }
    }
    float r0 = 1.0f / (d0 + 1e-8f);
    float r1 = 1.0f / (d1 + 1e-8f);
    float r2 = 1.0f / (d2 + 1e-8f);
    float rs = 1.0f / (r0 + r1 + r2);
    float w0 = r0 * rs, w1 = r1 * rs, w2 = r2 * rs;

    const float* p2 = pts2 + (size_t)b * SPTS * D2C;
    const float* q0 = p2 + (size_t)i0 * D2C;
    const float* q1 = p2 + (size_t)i1 * D2C;
    const float* q2 = p2 + (size_t)i2 * D2C;

    const size_t col = (size_t)b * NPTS + p;
    unsigned short* o = X0 + col * C0;
    const float* p1 = pts1 + ((size_t)b * NPTS + p) * D1C;
    for (int c = 0; c < D1C; ++c) o[c] = f2bf(p1[c]);
    for (int c = 0; c < D2C; ++c)
        o[D1C + c] = f2bf(w0 * q0[c] + w1 * q1[c] + w2 * q2[c]);
}

// ---------------------------------------------------------------------------
// BF16 WMMA GEMM:  Y[col][rowoff + m] = sum_k W[m][k] * X[col][k] + bias[m]
// W: [M][K] row-major bf16, X: [col][K] bf16 (feature contiguous).
// Block = 256 threads = 8 waves; wave tile = 16(M) x 64(N); block = 32 x 256.
// K is a compile-time constant: the K loop fully unrolls, every load folds
// its k-offset into the b128 immediate (K*2 <= 1536 B < 24-bit range), and
// the scheduler pipelines loads across WMMAs with partial loadcnt waits.
// Optional per-channel sum / sum-of-squares accumulation for BatchNorm.
// ---------------------------------------------------------------------------
template<int K>
__global__ __launch_bounds__(256)
void gemm_bf16_kernel(const unsigned short* __restrict__ W,
                      const unsigned short* __restrict__ X,
                      const float* __restrict__ bias,
                      unsigned short* __restrict__ Y,
                      int M, int NC, int ldY, int rowoff,
                      float* __restrict__ ssum, float* __restrict__ ssq) {
    union Frag { v4u q[2]; v16bf v; };

    const int tid  = threadIdx.x;
    const int wave = tid >> 5;
    const int lane = tid & 31;
    const int hlf  = lane >> 4;        // 0: lanes 0-15, 1: lanes 16-31
    const int l15  = lane & 15;
    const int wm   = wave >> 2;        // 0..1
    const int wn   = wave & 3;         // 0..3
    const int m0   = blockIdx.y * 32 + wm * 16;
    const int n0   = blockIdx.x * 256 + wn * 64;

    // A fragment source: row m, contiguous K chunk [k+16h .. k+16h+15]
    const unsigned short* wrow = W + (size_t)(m0 + l15) * K + hlf * 16;
    // B fragment sources: 4 N sub-tiles, contiguous K chunk [k+16h .. k+16h+15]
    const unsigned short* xcol[4];
    int cols[4];
#pragma unroll
    for (int t = 0; t < 4; ++t) {
        cols[t] = n0 + t * 16 + l15;
        xcol[t] = X + (size_t)cols[t] * K + hlf * 16;
    }

    v8f acc[4];
#pragma unroll
    for (int t = 0; t < 4; ++t)
#pragma unroll
        for (int r = 0; r < 8; ++r) acc[t][r] = 0.0f;

    // Fully-unrolled, software-pipelined K loop (K/32 steps).
    Frag a, b0, b1, b2, b3;
    {
        const v4u* pa = (const v4u*)(wrow);
        a.q[0] = pa[0]; a.q[1] = pa[1];
        const v4u* p0 = (const v4u*)(xcol[0]); b0.q[0] = p0[0]; b0.q[1] = p0[1];
        const v4u* p1 = (const v4u*)(xcol[1]); b1.q[0] = p1[0]; b1.q[1] = p1[1];
        const v4u* p2 = (const v4u*)(xcol[2]); b2.q[0] = p2[0]; b2.q[1] = p2[1];
        const v4u* p3 = (const v4u*)(xcol[3]); b3.q[0] = p3[0]; b3.q[1] = p3[1];
    }
#pragma unroll
    for (int k = 0; k < K; k += 32) {
        const int kn = (k + 32 < K) ? (k + 32) : 0;     // constant-folded
        Frag an, c0, c1, c2, c3;
        {
            const v4u* pa = (const v4u*)(wrow + kn);
            an.q[0] = pa[0]; an.q[1] = pa[1];
            const v4u* p0 = (const v4u*)(xcol[0] + kn); c0.q[0] = p0[0]; c0.q[1] = p0[1];
            const v4u* p1 = (const v4u*)(xcol[1] + kn); c1.q[0] = p1[0]; c1.q[1] = p1[1];
            const v4u* p2 = (const v4u*)(xcol[2] + kn); c2.q[0] = p2[0]; c2.q[1] = p2[1];
            const v4u* p3 = (const v4u*)(xcol[3] + kn); c3.q[0] = p3[0]; c3.q[1] = p3[1];
        }
        acc[0] = __builtin_amdgcn_wmma_f32_16x16x32_bf16(
            false, a.v, false, b0.v, (short)0, acc[0], false, false);
        acc[1] = __builtin_amdgcn_wmma_f32_16x16x32_bf16(
            false, a.v, false, b1.v, (short)0, acc[1], false, false);
        acc[2] = __builtin_amdgcn_wmma_f32_16x16x32_bf16(
            false, a.v, false, b2.v, (short)0, acc[2], false, false);
        acc[3] = __builtin_amdgcn_wmma_f32_16x16x32_bf16(
            false, a.v, false, b3.v, (short)0, acc[3], false, false);
        a = an; b0 = c0; b1 = c1; b2 = c2; b3 = c3;
    }

    // Epilogue: bias, store bf16 (X^T layout), optional BN stats.
#pragma unroll
    for (int r = 0; r < 8; ++r) {
        const int m = m0 + r + hlf * 8;      // C/D layout: vgpr r, half h -> M
        const float bm = bias[m];
        float s = 0.0f, q = 0.0f;
#pragma unroll
        for (int t = 0; t < 4; ++t) {
            float v = acc[t][r] + bm;
            Y[(size_t)cols[t] * ldY + rowoff + m] = f2bf(v);
            s += v;
            q += v * v;
        }
        if (ssum != nullptr) {
            // reduce the 16 lanes that share channel m (xor within half-group)
            s += __shfl_xor(s, 1); s += __shfl_xor(s, 2);
            s += __shfl_xor(s, 4); s += __shfl_xor(s, 8);
            q += __shfl_xor(q, 1); q += __shfl_xor(q, 2);
            q += __shfl_xor(q, 4); q += __shfl_xor(q, 8);
            if (l15 == 0) {
                atomicAdd(&ssum[m], s);
                atomicAdd(&ssq[m],  q);
            }
        }
    }
}

// ---------------------------------------------------------------------------
// Zero the BN stats accumulators
// ---------------------------------------------------------------------------
__global__ void zero2_kernel(float* a, float* b, int n) {
    int i = blockIdx.x * blockDim.x + threadIdx.x;
    if (i < n) { a[i] = 0.0f; b[i] = 0.0f; }
}

// ---------------------------------------------------------------------------
// BatchNorm(train) + ReLU, bf16 -> bf16.  Y layout [col][C].
// ---------------------------------------------------------------------------
__global__ void bn_relu_kernel(const unsigned short* __restrict__ Y,
                               unsigned short* __restrict__ Xo,
                               const float* __restrict__ ssum,
                               const float* __restrict__ ssq,
                               const float* __restrict__ gam,
                               const float* __restrict__ bet,
                               int C, int NC) {
    long long i = (long long)blockIdx.x * blockDim.x + threadIdx.x;
    long long tot = (long long)NC * C;
    if (i >= tot) return;
    int c = (int)(i % C);
    float cnt  = (float)NC;
    float mean = ssum[c] / cnt;
    float var  = ssq[c] / cnt - mean * mean;
    float inv  = rsqrtf(var + 1e-5f);
    float v    = bf2f(Y[i]);
    float o    = gam[c] * (v - mean) * inv + bet[c];
    Xo[i] = f2bf(fmaxf(o, 0.0f));
}

// ---------------------------------------------------------------------------
// fg = max over the n=8 columns of each group of Y2 (stored in X2[col][384:768]),
// broadcast into X2[col][0:384].
// ---------------------------------------------------------------------------
__global__ void fg_max_kernel(unsigned short* __restrict__ X2) {
    int i = blockIdx.x * blockDim.x + threadIdx.x;
    if (i >= (NCOL0 / NSUB) * C0) return;   // 4096 groups * 384 channels
    int c = i % C0;
    int g = i / C0;
    size_t base = (size_t)g * NSUB * C1;
    float m = -3.0e38f;
#pragma unroll
    for (int j = 0; j < NSUB; ++j)
        m = fmaxf(m, bf2f(X2[base + (size_t)j * C1 + C0 + c]));
    unsigned short h = f2bf(m);
#pragma unroll
    for (int j = 0; j < NSUB; ++j)
        X2[base + (size_t)j * C1 + c] = h;
}

// ---------------------------------------------------------------------------
// tok = max over n=8 columns of Y4 -> bf16 (GEMM input) + f32 (residual)
// ---------------------------------------------------------------------------
__global__ void tok_max_kernel(const unsigned short* __restrict__ Y4,
                               unsigned short* __restrict__ T0,
                               float* __restrict__ Tf) {
    int i = blockIdx.x * blockDim.x + threadIdx.x;
    if (i >= NCOLG * OC) return;
    int c  = i % OC;
    int bg = i / OC;
    size_t base = (size_t)bg * NSUB * OC;
    float m = -3.0e38f;
#pragma unroll
    for (int j = 0; j < NSUB; ++j)
        m = fmaxf(m, bf2f(Y4[base + (size_t)j * OC + c]));
    T0[i] = f2bf(m);
    Tf[i] = m;
}

// ---------------------------------------------------------------------------
// Final: relu( relu(bn(Y6)) + tok ) -> f32 out [B,G,out_c]
// ---------------------------------------------------------------------------
__global__ void final_kernel(const unsigned short* __restrict__ Y6,
                             const float* __restrict__ ssum,
                             const float* __restrict__ ssq,
                             const float* __restrict__ gam,
                             const float* __restrict__ bet,
                             const float* __restrict__ Tf,
                             float* __restrict__ out) {
    int i = blockIdx.x * blockDim.x + threadIdx.x;
    if (i >= NCOLG * OC) return;
    int c = i % OC;
    float cnt  = (float)NCOLG;
    float mean = ssum[c] / cnt;
    float var  = ssq[c] / cnt - mean * mean;
    float inv  = rsqrtf(var + 1e-5f);
    float v    = bf2f(Y6[i]);
    float h    = fmaxf(gam[c] * (v - mean) * inv + bet[c], 0.0f);
    out[i] = fmaxf(h + Tf[i], 0.0f);
}

// ---------------------------------------------------------------------------
// Host-side orchestration
// ---------------------------------------------------------------------------
extern "C" void kernel_launch(void* const* d_in, const int* in_sizes, int n_in,
                              void* d_out, int out_size, void* d_ws, size_t ws_size,
                              hipStream_t stream) {
    const float* xyz1   = (const float*)d_in[0];
    const float* xyz2   = (const float*)d_in[1];
    const float* pts1   = (const float*)d_in[2];
    const float* pts2   = (const float*)d_in[3];
    const float* fc1_w  = (const float*)d_in[4];
    const float* fc1_b  = (const float*)d_in[5];
    const float* bn1_g  = (const float*)d_in[6];
    const float* bn1_b  = (const float*)d_in[7];
    const float* fc2_w  = (const float*)d_in[8];
    const float* fc2_b  = (const float*)d_in[9];
    const float* sc1_w  = (const float*)d_in[10];
    const float* sc1_b  = (const float*)d_in[11];
    const float* bn2_g  = (const float*)d_in[12];
    const float* bn2_b  = (const float*)d_in[13];
    const float* sc2_w  = (const float*)d_in[14];
    const float* sc2_b  = (const float*)d_in[15];
    const float* e1_w   = (const float*)d_in[16];
    const float* e1_b   = (const float*)d_in[17];
    const float* ebn1_g = (const float*)d_in[18];
    const float* ebn1_b = (const float*)d_in[19];
    const float* e2_w   = (const float*)d_in[20];
    const float* e2_b   = (const float*)d_in[21];
    const float* ebn2_g = (const float*)d_in[22];
    const float* ebn2_b = (const float*)d_in[23];
    float* out = (float*)d_out;

    // ---- workspace layout (bytes; 256-aligned) ------------------------------
    char* base = (char*)d_ws;
    size_t off = 0;
    auto alloc = [&](size_t bytes) -> char* {
        char* p = base + off;
        off += (bytes + 255) & ~(size_t)255;
        return p;
    };
    unsigned short* wfc1 = (unsigned short*)alloc((size_t)C0 * C0 * 2);
    unsigned short* wfc2 = (unsigned short*)alloc((size_t)C0 * C0 * 2);
    unsigned short* wsc1 = (unsigned short*)alloc((size_t)OC * C1 * 2);
    unsigned short* wsc2 = (unsigned short*)alloc((size_t)OC * OC * 2);
    unsigned short* we1  = (unsigned short*)alloc((size_t)OC * OC * 2);
    unsigned short* we2  = (unsigned short*)alloc((size_t)OC * OC * 2);
    unsigned short* XA   = (unsigned short*)alloc((size_t)NCOL0 * C1 * 2); // 50.3 MB
    unsigned short* XB   = (unsigned short*)alloc((size_t)NCOL0 * C1 * 2); // 50.3 MB
    unsigned short* T0   = (unsigned short*)alloc((size_t)NCOLG * OC * 2); // 4 MB
    float*          Tf   = (float*)alloc((size_t)NCOLG * OC * 4);          // 8 MB
    float*          ssum = (float*)alloc(1024 * 4);
    float*          ssq  = (float*)alloc(1024 * 4);
    (void)ws_size; (void)in_sizes; (void)n_in; (void)out_size;

    // ---- 0) weights -> bf16 -------------------------------------------------
    auto cvt = [&](const float* s, unsigned short* d, int n) {
        cvt_w_kernel<<<(n + 255) / 256, 256, 0, stream>>>(s, d, n);
    };
    cvt(fc1_w, wfc1, C0 * C0);
    cvt(fc2_w, wfc2, C0 * C0);
    cvt(sc1_w, wsc1, OC * C1);
    cvt(sc2_w, wsc2, OC * OC);
    cvt(e1_w,  we1,  OC * OC);
    cvt(e2_w,  we2,  OC * OC);

    // ---- 1) 3-NN interpolate + concat -> X0 in XA [32768][384] -------------
    knn_interp_kernel<<<dim3(NPTS / 128, BATCH), 128, 0, stream>>>(
        xyz1, xyz2, pts1, pts2, XA);

    // ---- 2) fc1 GEMM (+stats) : Y1 -> XB [col][384] -------------------------
    zero2_kernel<<<4, 256, 0, stream>>>(ssum, ssq, 1024);
    gemm_bf16_kernel<C0><<<dim3(NCOL0 / 256, C0 / 32), 256, 0, stream>>>(
        wfc1, XA, fc1_b, XB, C0, NCOL0, C0, 0, ssum, ssq);
    // bn1 + relu : XB -> XA (X1)
    {
        long long tot = (long long)NCOL0 * C0;
        bn_relu_kernel<<<(unsigned)((tot + 255) / 256), 256, 0, stream>>>(
            XB, XA, ssum, ssq, bn1_g, bn1_b, C0, NCOL0);
    }

    // ---- 3) fc2 GEMM (raw) : Y2 -> XB as X2[col][768], rows 384..767 --------
    gemm_bf16_kernel<C0><<<dim3(NCOL0 / 256, C0 / 32), 256, 0, stream>>>(
        wfc2, XA, fc2_b, XB, C0, NCOL0, C1, C0, nullptr, nullptr);
    // fg broadcast into X2 rows 0..383
    fg_max_kernel<<<((NCOL0 / NSUB) * C0 + 255) / 256, 256, 0, stream>>>(XB);

    // ---- 4) sc1 GEMM (+stats) : Y3 -> XA [col][512] -------------------------
    zero2_kernel<<<4, 256, 0, stream>>>(ssum, ssq, 1024);
    gemm_bf16_kernel<C1><<<dim3(NCOL0 / 256, OC / 32), 256, 0, stream>>>(
        wsc1, XB, sc1_b, XA, OC, NCOL0, OC, 0, ssum, ssq);
    // bn2 + relu : XA -> XB (X3)
    {
        long long tot = (long long)NCOL0 * OC;
        bn_relu_kernel<<<(unsigned)((tot + 255) / 256), 256, 0, stream>>>(
            XA, XB, ssum, ssq, bn2_g, bn2_b, OC, NCOL0);
    }

    // ---- 5) sc2 GEMM (raw) : Y4 -> XA [col][512] ----------------------------
    gemm_bf16_kernel<OC><<<dim3(NCOL0 / 256, OC / 32), 256, 0, stream>>>(
        wsc2, XB, sc2_b, XA, OC, NCOL0, OC, 0, nullptr, nullptr);
    // tok = group max -> T0 (bf16) + Tf (f32 residual)
    tok_max_kernel<<<(NCOLG * OC + 255) / 256, 256, 0, stream>>>(XA, T0, Tf);

    // ---- 6) e1 GEMM (+stats, 4096 cols) : Y5 -> XA --------------------------
    zero2_kernel<<<4, 256, 0, stream>>>(ssum, ssq, 1024);
    gemm_bf16_kernel<OC><<<dim3(NCOLG / 256, OC / 32), 256, 0, stream>>>(
        we1, T0, e1_b, XA, OC, NCOLG, OC, 0, ssum, ssq);
    bn_relu_kernel<<<(NCOLG * OC + 255) / 256, 256, 0, stream>>>(
        XA, XB, ssum, ssq, ebn1_g, ebn1_b, OC, NCOLG);

    // ---- 7) e2 GEMM (+stats) : Y6 -> XA -------------------------------------
    zero2_kernel<<<4, 256, 0, stream>>>(ssum, ssq, 1024);
    gemm_bf16_kernel<OC><<<dim3(NCOLG / 256, OC / 32), 256, 0, stream>>>(
        we2, XB, e2_b, XA, OC, NCOLG, OC, 0, ssum, ssq);

    // ---- 8) final bn + relu + residual + relu -> d_out ----------------------
    final_kernel<<<(NCOLG * OC + 255) / 256, 256, 0, stream>>>(
        XA, ssum, ssq, ebn2_g, ebn2_b, Tf, out);
}
